// DetectionLoss_9216999817597
// MI455X (gfx1250) — matrix operand order repaired and checked by reading, not compiled, not run on hardware
//
#include <hip/hip_runtime.h>

// Problem sizes fixed by the reference setup_inputs().
constexpr int kB = 64;     // batch
constexpr int kQ = 900;    // queries
constexpr int kG = 100;    // max GT boxes
constexpr int kC1 = 11;    // NUM_CLASSES + 1
constexpr int NT = 256;    // threads for reduction-style kernels

typedef __attribute__((ext_vector_type(16))) _Float16 v16h;
typedef __attribute__((ext_vector_type(8)))  float    v8f;

// ---------------------------------------------------------------------------
// helpers
// ---------------------------------------------------------------------------
__device__ __forceinline__ float giou_pair(float4 p, float4 g) {
    // cxcywh -> xyxy, then GIoU exactly as the reference (clip eps = 1e-6).
    float px0 = p.x - 0.5f * p.z, py0 = p.y - 0.5f * p.w;
    float px1 = p.x + 0.5f * p.z, py1 = p.y + 0.5f * p.w;
    float gx0 = g.x - 0.5f * g.z, gy0 = g.y - 0.5f * g.w;
    float gx1 = g.x + 0.5f * g.z, gy1 = g.y + 0.5f * g.w;
    float a1 = fmaxf(px1 - px0, 0.f) * fmaxf(py1 - py0, 0.f);
    float a2 = fmaxf(gx1 - gx0, 0.f) * fmaxf(gy1 - gy0, 0.f);
    float iw = fmaxf(fminf(px1, gx1) - fmaxf(px0, gx0), 0.f);
    float ih = fmaxf(fminf(py1, gy1) - fmaxf(py0, gy0), 0.f);
    float inter = iw * ih;
    float uni = a1 + a2 - inter;
    float iou = inter / fmaxf(uni, 1e-6f);
    float ew = fmaxf(fmaxf(px1, gx1) - fminf(px0, gx0), 0.f);
    float eh = fmaxf(fmaxf(py1, gy1) - fminf(py0, gy0), 0.f);
    float enc = ew * eh;
    return iou - (enc - uni) / fmaxf(enc, 1e-6f);
}

__device__ __forceinline__ float l1_pair(float4 p, float4 g) {
    return fabsf(p.x - g.x) + fabsf(p.y - g.y) + fabsf(p.z - g.z) + fabsf(p.w - g.w);
}

// ---------------------------------------------------------------------------
// Kernel 1: softmax(pred_logits) -> f16 probabilities, padded to 16 classes.
// ---------------------------------------------------------------------------
__global__ void softmax_f16_kernel(const float* __restrict__ logits,
                                   _Float16* __restrict__ prob16) {
    int idx = blockIdx.x * blockDim.x + threadIdx.x;
    if (idx >= kB * kQ) return;
    const float* x = logits + (size_t)idx * kC1;
    float mx = x[0];
#pragma unroll
    for (int c = 1; c < kC1; ++c) mx = fmaxf(mx, x[c]);
    float e[kC1]; float s = 0.f;
#pragma unroll
    for (int c = 0; c < kC1; ++c) { e[c] = __expf(x[c] - mx); s += e[c]; }
    float inv = 1.f / s;
    _Float16* o = prob16 + (size_t)idx * 16;
#pragma unroll
    for (int c = 0; c < kC1; ++c) o[c] = (_Float16)(e[c] * inv);
#pragma unroll
    for (int c = kC1; c < 16; ++c) o[c] = (_Float16)0.f;
}

// ---------------------------------------------------------------------------
// Kernel 2: transposed cost  costT[b,g,q] = 2*(-prob.onehot) + 5*L1 - 2*GIoU.
// One wave per 16x16 (q,g) tile; class term via v_wmma_f32_16x16x32_f16.
// Transposed (G-major) storage makes the Hungarian column scans contiguous.
// ---------------------------------------------------------------------------
__global__ __launch_bounds__(32) void cost_tile_kernel(
    const _Float16* __restrict__ prob16,
    const float* __restrict__ pred_boxes,
    const float* __restrict__ gt_boxes,
    const int* __restrict__ gt_classes,
    float* __restrict__ costT) {
    const int lane = threadIdx.x;            // 0..31
    const int b  = blockIdx.z;
    const int q0 = blockIdx.x * 16;
    const int g0 = blockIdx.y * 16;
    const bool lo = lane < 16;
    const int row = lane & 15;

    // ---- A matrix: 16x32 f16 (prob rows, K padded with zeros) ----
    // lanes 0-15: halves 0..7 = K 0..7 ; lanes 16-31: halves 0..7 = K 8..15
    v16h a = {};
    {
        int q = q0 + row;
        if (q < kQ) {
            const _Float16* src = prob16 + ((size_t)b * kQ + q) * 16 + (lo ? 0 : 8);
#pragma unroll
            for (int j = 0; j < 8; ++j) a[j] = src[j];
        }
    }
    // ---- B matrix: 32x16 f16 one-hot of clipped GT class ----
    // lanes 0-15: halves 0..7 = K 0..7 ; lanes 16-31: halves 0..7 = K 8..15; col N = lane&15.
    v16h bb = {};
    int gcolumn = g0 + row;
    if (gcolumn < kG) {
        int cls = gt_classes[b * kG + gcolumn];
        cls = cls < 0 ? 0 : (cls > 9 ? 9 : cls);
        if ((cls >> 3) == (lo ? 0 : 1)) bb[cls & 7] = (_Float16)1.0f;
    }
    v8f c = {};
    v8f d = __builtin_amdgcn_wmma_f32_16x16x32_f16(
        false, a, false, bb, (short)0, c, false, false);

    // ---- finish each of this lane's 8 C entries with L1 + GIoU ----
    const int N = lane & 15;
    const int g = g0 + N;
    float4 gbox = make_float4(0.f, 0.f, 0.f, 0.f);
    if (g < kG) gbox = ((const float4*)gt_boxes)[(size_t)b * kG + g];
#pragma unroll
    for (int r = 0; r < 8; ++r) {
        int M = r + (lo ? 0 : 8);
        int q = q0 + M;
        if (q < kQ && g < kG) {
            float4 pbox = ((const float4*)pred_boxes)[(size_t)b * kQ + q];
            float l1 = l1_pair(pbox, gbox);
            float gi = giou_pair(pbox, gbox);
            float val = 2.0f * (-d[r]) + 5.0f * l1 - 2.0f * gi;
            costT[((size_t)b * kG + g) * kQ + q] = val;   // G-major (transposed)
        }
    }
}

// ---------------------------------------------------------------------------
// Kernel 3: Jonker-Volgenant assignment, one workgroup per batch image.
// Rows = valid GT (n<=100), cols = queries (m=900). f64 duals (matches ref).
// costT is (B,G,Q): a row scan is a contiguous 3.6KB stream (L2-resident).
// match[b*kQ+q] = assigned gt index, or -1.
// ---------------------------------------------------------------------------
__global__ __launch_bounds__(NT) void hungarian_kernel(
    const float* __restrict__ costT,
    const int* __restrict__ gt_classes,
    int* __restrict__ match) {
    const int b = blockIdx.x;
    const int tid = threadIdx.x;
    const int m = kQ;

    __shared__ double s_v[kQ + 1];
    __shared__ double s_minv[kQ + 1];
    __shared__ double s_u[kG + 1];
    __shared__ int    s_p[kQ + 1];
    __shared__ int    s_way[kQ + 1];
    __shared__ unsigned char s_used[kQ + 1];
    __shared__ int    s_vidx[kG];
    __shared__ int    s_n, s_j0;
    __shared__ double red_val[NT];
    __shared__ int    red_idx[NT];

    if (tid == 0) {
        int n = 0;
        for (int g = 0; g < kG; ++g)
            if (gt_classes[b * kG + g] >= 0) s_vidx[n++] = g;
        s_n = n;
    }
    for (int j = tid; j <= m; j += NT) { s_v[j] = 0.0; s_p[j] = 0; }
    for (int i = tid; i <= kG; i += NT) s_u[i] = 0.0;
    __syncthreads();
    const int n = s_n;
    const float* Cbt = costT + (size_t)b * kG * kQ;

    for (int i = 1; i <= n; ++i) {
        if (tid == 0) { s_p[0] = i; s_j0 = 0; }
        for (int j = tid; j <= m; j += NT) {
            s_minv[j] = 1e300; s_way[j] = 0; s_used[j] = 0;
        }
        __syncthreads();
        while (true) {
            const int j0 = s_j0;
            if (tid == 0) s_used[j0] = 1;
            __syncthreads();
            const int i0 = s_p[j0];
            const double ui0 = s_u[i0];
            const float* Crow = Cbt + (size_t)s_vidx[i0 - 1] * kQ;  // contiguous
            double bestv = 1e300; int bestj = -1;
            for (int j = tid + 1; j <= m; j += NT) {
                if (j + NT <= m)  // pull next contiguous chunk into cache
                    __builtin_prefetch(&Crow[j - 1 + NT], 0, 1);
                if (!s_used[j]) {
                    double cur = (double)Crow[j - 1] - ui0 - s_v[j];
                    if (cur < s_minv[j]) { s_minv[j] = cur; s_way[j] = j0; }
                    double mv = s_minv[j];
                    if (mv < bestv) { bestv = mv; bestj = j; }
                }
            }
            red_val[tid] = bestv; red_idx[tid] = bestj;
            __syncthreads();
            for (int s = NT / 2; s > 0; s >>= 1) {
                if (tid < s) {
                    double ov = red_val[tid + s]; int oj = red_idx[tid + s];
                    double mv = red_val[tid];     int mj = red_idx[tid];
                    if (oj >= 0 &&
                        (mj < 0 || ov < mv || (ov == mv && oj < mj))) {
                        red_val[tid] = ov; red_idx[tid] = oj;
                    }
                }
                __syncthreads();
            }
            const int j1 = red_idx[0];
            const double delta = red_val[0];
            if (tid == 0) s_j0 = j1;
            for (int j = tid; j <= m; j += NT) {
                if (s_used[j]) { s_u[s_p[j]] += delta; s_v[j] -= delta; }
                else            s_minv[j] -= delta;
            }
            __syncthreads();
            if (s_p[j1] == 0) break;  // uniform across block
        }
        if (tid == 0) {               // augment along the recorded path
            int j0 = s_j0;
            while (j0 != 0) { int jn = s_way[j0]; s_p[j0] = s_p[jn]; j0 = jn; }
        }
        __syncthreads();
    }
    for (int j = tid + 1; j <= m; j += NT)
        match[(size_t)b * kQ + (j - 1)] = (s_p[j] > 0) ? s_vidx[s_p[j] - 1] : -1;
}

// ---------------------------------------------------------------------------
// Kernel 4: per-query loss contributions, deterministic block tree reduction.
// partials[blk*5 + {0..4}] = {sum wt*nll, sum wt, sum L1, sum (1-giou), n}
// ---------------------------------------------------------------------------
__global__ __launch_bounds__(NT) void loss_partial_kernel(
    const float* __restrict__ logits,
    const float* __restrict__ pred_boxes,
    const float* __restrict__ gt_boxes,
    const int* __restrict__ gt_classes,
    const int* __restrict__ match,
    float* __restrict__ partials) {
    __shared__ float sh[5][NT];
    const int tid = threadIdx.x;
    const int idx = blockIdx.x * NT + tid;
    float acc[5] = {0.f, 0.f, 0.f, 0.f, 0.f};
    if (idx < kB * kQ) {
        const int b = idx / kQ;
        const float* x = logits + (size_t)idx * kC1;
        float xr[kC1];
        float mx = -3.4e38f;
#pragma unroll
        for (int cc = 0; cc < kC1; ++cc) { xr[cc] = x[cc]; mx = fmaxf(mx, xr[cc]); }
        float s = 0.f;
#pragma unroll
        for (int cc = 0; cc < kC1; ++cc) s += __expf(xr[cc] - mx);
        float lse = __logf(s) + mx;
        int g = match[idx];
        int t = 10;
        if (g >= 0) t = gt_classes[b * kG + g];
        float wt = (t == 10) ? 0.1f : 1.0f;
        acc[0] = wt * (lse - xr[t]);
        acc[1] = wt;
        if (g >= 0) {
            float4 mp = ((const float4*)pred_boxes)[idx];
            float4 mg = ((const float4*)gt_boxes)[(size_t)b * kG + g];
            acc[2] = l1_pair(mp, mg);
            acc[3] = 1.0f - giou_pair(mp, mg);
            acc[4] = 1.0f;
        }
    }
#pragma unroll
    for (int k = 0; k < 5; ++k) sh[k][tid] = acc[k];
    __syncthreads();
    for (int s = NT / 2; s > 0; s >>= 1) {
        if (tid < s)
#pragma unroll
            for (int k = 0; k < 5; ++k) sh[k][tid] += sh[k][tid + s];
        __syncthreads();
    }
    if (tid == 0)
#pragma unroll
        for (int k = 0; k < 5; ++k) partials[blockIdx.x * 5 + k] = sh[k][0];
}

__global__ void finalize_kernel(const float* __restrict__ partials, int nblocks,
                                float* __restrict__ out) {
    if (blockIdx.x == 0 && threadIdx.x == 0) {
        float a[5] = {0.f, 0.f, 0.f, 0.f, 0.f};
        for (int i = 0; i < nblocks; ++i)
            for (int k = 0; k < 5; ++k) a[k] += partials[i * 5 + k];
        float nm = fmaxf(a[4], 1.0f);
        float cls = a[0] / a[1];
        out[0] = 2.0f * cls + 5.0f * a[2] / nm + 2.0f * a[3] / nm;
    }
}

// ---------------------------------------------------------------------------
// launcher
// ---------------------------------------------------------------------------
extern "C" void kernel_launch(void* const* d_in, const int* in_sizes, int n_in,
                              void* d_out, int out_size, void* d_ws, size_t ws_size,
                              hipStream_t stream) {
    const float* pred_logits = (const float*)d_in[0];
    const float* pred_boxes  = (const float*)d_in[1];
    const int*   gt_classes  = (const int*)d_in[2];
    const float* gt_boxes    = (const float*)d_in[3];
    float* out = (float*)d_out;

    char* ws = (char*)d_ws;
    size_t off = 0;
    auto take = [&](size_t bytes) {
        void* p = ws + off;
        off = (off + bytes + 255) & ~(size_t)255;
        return p;
    };
    _Float16* prob16 = (_Float16*)take((size_t)kB * kQ * 16 * sizeof(_Float16));
    float*    costT  = (float*)take((size_t)kB * kG * kQ * sizeof(float));
    int*      match  = (int*)take((size_t)kB * kQ * sizeof(int));
    const int nblocks = (kB * kQ + NT - 1) / NT;          // 225
    float*    partials = (float*)take((size_t)nblocks * 5 * sizeof(float));
    (void)ws_size; (void)in_sizes; (void)n_in; (void)out_size;

    softmax_f16_kernel<<<nblocks, NT, 0, stream>>>(pred_logits, prob16);

    dim3 cgrid((kQ + 15) / 16, (kG + 15) / 16, kB);        // 57 x 7 x 64
    cost_tile_kernel<<<cgrid, 32, 0, stream>>>(prob16, pred_boxes, gt_boxes,
                                               gt_classes, costT);

    hungarian_kernel<<<kB, NT, 0, stream>>>(costT, gt_classes, match);

    loss_partial_kernel<<<nblocks, NT, 0, stream>>>(pred_logits, pred_boxes,
                                                    gt_boxes, gt_classes, match,
                                                    partials);
    finalize_kernel<<<1, 64, 0, stream>>>(partials, nblocks, out);
}
